// ContextualLoss3D_patches_46583215293138
// MI455X (gfx1250) — matrix operand order repaired and checked by reading, not compiled, not run on hardware
//
#include <hip/hip_runtime.h>
#include <hip/hip_bf16.h>
#include <math.h>

typedef __attribute__((ext_vector_type(16))) _Float16 v16h;
typedef __attribute__((ext_vector_type(8)))  _Float16 v8h;
typedef __attribute__((ext_vector_type(8)))  float    v8f;

#define NPATCH      216          // (48/8)^3
#define NCH         32
#define MPOS        512          // 8^3
#define CH_STRIDE   110592       // 48*48*48
#define H_STRIDE    2304         // 48*48
#define W_STRIDE    48
#define LDS_ROW     40           // padded row stride in halves (80B, 16B-aligned, conflict-free)
#define CXEPS       1e-5f
#define CXNORM_EPS  1e-12f

union FragH { v16h v; v8h h[2]; };

__global__ __launch_bounds__(256)
void cx_patch_kernel(const float* __restrict__ x, const float* __restrict__ y,
                     float* __restrict__ maxcx /* [2][216][512] */)
{
    __shared__ _Float16 xnT[MPOS * LDS_ROW];   // [m][c] transposed, f16
    __shared__ _Float16 ynT[MPOS * LDS_ROW];
    __shared__ float    dminS[MPOS];
    __shared__ float    sS[MPOS];
    __shared__ float    muS[NCH];
    __shared__ float    red[NCH * 8];

    const int p    = blockIdx.x;
    const int n    = blockIdx.y;
    const int t    = threadIdx.x;
    const int lane = t & 31;
    const int wave = t >> 5;

    const int ph = p / 36;
    const int pw = (p / 6) % 6;
    const int pd = p % 6;
    const int sp_base = (ph * 8) * H_STRIDE + (pw * 8) * W_STRIDE + (pd * 8);

    // ---- y_mu: mean over both batches and all 512 spatial positions, per channel ----
    {
        const int c = t & 31;
        const int g = t >> 5;                 // 8 groups of 32 threads
        float acc = 0.f;
        for (int nn = 0; nn < 2; ++nn) {
            const float* yc = y + (size_t)(nn * NCH + c) * CH_STRIDE + sp_base;
            for (int m = g; m < MPOS; m += 8) {
                int dh = m >> 6, dw = (m >> 3) & 7, dd = m & 7;
                acc += yc[dh * H_STRIDE + dw * W_STRIDE + dd];
            }
        }
        red[c * 8 + g] = acc;
    }
    __syncthreads();
    if (t < NCH) {
        float tot = 0.f;
        for (int g = 0; g < 8; ++g) tot += red[t * 8 + g];
        muS[t] = tot * (1.f / 1024.f);
    }
    __syncthreads();

    // ---- center, L2-normalize along channels, store f16 transposed [m][c] ----
    for (int m = t; m < MPOS; m += 256) {
        int dh = m >> 6, dw = (m >> 3) & 7, dd = m & 7;
        int sp = sp_base + dh * H_STRIDE + dw * W_STRIDE + dd;
        float vx[NCH], vy[NCH];
        float ssx = 0.f, ssy = 0.f;
#pragma unroll
        for (int c = 0; c < NCH; ++c) {
            float mu = muS[c];
            float a = x[(size_t)(n * NCH + c) * CH_STRIDE + sp] - mu;
            float b = y[(size_t)(n * NCH + c) * CH_STRIDE + sp] - mu;
            vx[c] = a; vy[c] = b;
            ssx += a * a; ssy += b * b;
        }
        float ix = 1.f / fmaxf(sqrtf(ssx), CXNORM_EPS);
        float iy = 1.f / fmaxf(sqrtf(ssy), CXNORM_EPS);
#pragma unroll
        for (int c = 0; c < NCH; ++c) {
            xnT[m * LDS_ROW + c] = (_Float16)(vx[c] * ix);
            ynT[m * LDS_ROW + c] = (_Float16)(vy[c] * iy);
        }
    }
    __syncthreads();

    // Fragment addressing (this code's contiguous k order for A 16x32 / B 32x16, not the ISA table order):
    // A: lane<16 holds M=lane, K=0..15        ; lane>=16 holds M=lane-16, K=16..31
    // B: lane<16 holds N=lane, K=0..15        ; lane>=16 holds N=lane-16, K=16..31
    const int halfSel = (lane >= 16) ? 1 : 0;
    const int row16   = lane & 15;
    const int ka      = halfSel ? 16 : 0;
    const int kb      = halfSel ? 16 : 0;

    // ---- fused pass A (row mins) + pass B (row exp-sums); wave owns i-tiles ----
    for (int it = wave; it < 32; it += 8) {
        FragH a;
        const _Float16* ar = &xnT[(it * 16 + row16) * LDS_ROW];
        a.h[0] = *(const v8h*)(ar + ka);
        a.h[1] = *(const v8h*)(ar + ka + 8);

        float pmin[8];
#pragma unroll
        for (int r = 0; r < 8; ++r) pmin[r] = 3.0e38f;

        for (int jt = 0; jt < 32; ++jt) {
            FragH b;
            const _Float16* br = &ynT[(jt * 16 + row16) * LDS_ROW];
            b.h[0] = *(const v8h*)(br + kb);
            b.h[1] = *(const v8h*)(br + kb + 8);
            v8f c0 = {};
            v8f d = __builtin_amdgcn_wmma_f32_16x16x32_f16(
                false, a.v, false, b.v, (short)0, c0, false, false);
#pragma unroll
            for (int r = 0; r < 8; ++r) pmin[r] = fminf(pmin[r], 1.f - d[r]);
        }
#pragma unroll
        for (int mask = 1; mask <= 8; mask <<= 1)
#pragma unroll
            for (int r = 0; r < 8; ++r)
                pmin[r] = fminf(pmin[r], __shfl_xor(pmin[r], mask, 32));

        float psum[8];
#pragma unroll
        for (int r = 0; r < 8; ++r) psum[r] = 0.f;

        for (int jt = 0; jt < 32; ++jt) {
            FragH b;
            const _Float16* br = &ynT[(jt * 16 + row16) * LDS_ROW];
            b.h[0] = *(const v8h*)(br + kb);
            b.h[1] = *(const v8h*)(br + kb + 8);
            v8f c0 = {};
            v8f d = __builtin_amdgcn_wmma_f32_16x16x32_f16(
                false, a.v, false, b.v, (short)0, c0, false, false);
#pragma unroll
            for (int r = 0; r < 8; ++r)
                psum[r] += __expf(1.f - (1.f - d[r]) / (pmin[r] + CXEPS));
        }
#pragma unroll
        for (int mask = 1; mask <= 8; mask <<= 1)
#pragma unroll
            for (int r = 0; r < 8; ++r)
                psum[r] += __shfl_xor(psum[r], mask, 32);

        if (row16 == 0) {
            int rb = it * 16 + (halfSel ? 8 : 0);
#pragma unroll
            for (int r = 0; r < 8; ++r) {
                dminS[rb + r] = pmin[r];
                sS[rb + r]    = psum[r];
            }
        }
    }
    __syncthreads();

    // ---- pass C: column max of cx = w/s; wave owns j-tiles, loops i-tiles ----
    for (int jt = wave; jt < 32; jt += 8) {
        FragH b;
        const _Float16* br = &ynT[(jt * 16 + row16) * LDS_ROW];
        b.h[0] = *(const v8h*)(br + kb);
        b.h[1] = *(const v8h*)(br + kb + 8);

        float pmax = -3.0e38f;
        for (int it = 0; it < 32; ++it) {
            FragH a;
            const _Float16* ar = &xnT[(it * 16 + row16) * LDS_ROW];
            a.h[0] = *(const v8h*)(ar + ka);
            a.h[1] = *(const v8h*)(ar + ka + 8);
            v8f c0 = {};
            v8f d = __builtin_amdgcn_wmma_f32_16x16x32_f16(
                false, a.v, false, b.v, (short)0, c0, false, false);
            int rb = it * 16 + (halfSel ? 8 : 0);
            v8f dm = *(const v8f*)(&dminS[rb]);
            v8f sv = *(const v8f*)(&sS[rb]);
#pragma unroll
            for (int r = 0; r < 8; ++r) {
                float w = __expf(1.f - (1.f - d[r]) / (dm[r] + CXEPS));
                pmax = fmaxf(pmax, w / sv[r]);
            }
        }
        pmax = fmaxf(pmax, __shfl_xor(pmax, 16, 32));
        if (lane < 16)
            maxcx[((size_t)n * NPATCH + p) * MPOS + jt * 16 + lane] = pmax;
    }
}

__global__ __launch_bounds__(256)
void cx_finalize_kernel(const float* __restrict__ maxcx, float* __restrict__ out)
{
    __shared__ float red[256];
    const int t = threadIdx.x;
    float acc = 0.f;
    for (int idx = t; idx < 2 * MPOS; idx += 256) {
        int n = idx >> 9;
        int j = idx & (MPOS - 1);
        float s = 0.f;
        for (int p = 0; p < NPATCH; ++p)
            s += maxcx[((size_t)n * NPATCH + p) * MPOS + j];
        acc += -__logf(s * (1.f / (float)NPATCH) + CXEPS);
    }
    red[t] = acc;
    __syncthreads();
    for (int off = 128; off > 0; off >>= 1) {
        if (t < off) red[t] += red[t + off];
        __syncthreads();
    }
    if (t == 0) out[0] = red[0] * (1.f / 1024.f);
}

extern "C" void kernel_launch(void* const* d_in, const int* in_sizes, int n_in,
                              void* d_out, int out_size, void* d_ws, size_t ws_size,
                              hipStream_t stream) {
    const float* x = (const float*)d_in[0];
    const float* y = (const float*)d_in[1];
    float* out   = (float*)d_out;
    float* maxcx = (float*)d_ws;     // needs 2*216*512*4 = 884,736 bytes

    dim3 grid(NPATCH, 2);
    cx_patch_kernel<<<grid, 256, 0, stream>>>(x, y, maxcx);
    cx_finalize_kernel<<<1, 256, 0, stream>>>(maxcx, out);
}
